// MoE_2860448219291
// MI455X (gfx1250) — matrix operand order repaired and checked by reading, not compiled, not run on hardware
//
#include <hip/hip_runtime.h>
#include <math.h>

// ---------------- problem constants ----------------
#define TT 8192      // tokens = 4*2048
#define HH 2048      // hidden
#define II 2048      // intermediate
#define EE 8         // experts
#define KK_TOP 2

constexpr int BM = 128;   // token-tile
constexpr int BN = 128;   // out-col tile
constexpr int BK = 32;    // k step (f16 WMMA K=32)

typedef __attribute__((ext_vector_type(16))) _Float16 v16h;
typedef __attribute__((ext_vector_type(8)))  float    v8f;
typedef __attribute__((ext_vector_type(4)))  float    v4f;

// pack two f16 (k even, k odd) into one dword (low = even k)
static __device__ __forceinline__ unsigned int pk_f16(float a, float b) {
    union { _Float16 h[2]; unsigned int u; } v;
    v.h[0] = (_Float16)a; v.h[1] = (_Float16)b;
    return v.u;
}

static __device__ __forceinline__ float gelu_exact(float v) {
    return 0.5f * v * (1.0f + erff(v * 0.70710678118654752f));
}

// ---------------- kernel 0: init (vectorized zero of out + counters) ----------------
__global__ void moe_init_kernel(float* __restrict__ out, int n4, int* __restrict__ cnt) {
    if (blockIdx.x == 0 && threadIdx.x < EE) cnt[threadIdx.x] = 0;
    const v4f z = {0.f, 0.f, 0.f, 0.f};
    v4f* o4 = (v4f*)out;
    int stride = gridDim.x * blockDim.x;
    for (int i = blockIdx.x * blockDim.x + threadIdx.x; i < n4; i += stride)
        o4[i] = z;
}

// ---------------- kernel 1: gate (one wave32 per token) ----------------
__global__ __launch_bounds__(256)
void moe_gate_kernel(const float* __restrict__ x, const float* __restrict__ gw,
                     const float* __restrict__ alpha, _Float16* __restrict__ x_h,
                     int* __restrict__ cnt, int* __restrict__ expTok,
                     float* __restrict__ wTok) {
    const int wv = threadIdx.x >> 5;
    const int lane = threadIdx.x & 31;
    const int t = blockIdx.x * 8 + wv;
    if (t >= TT) return;

    const float* xr = x + (size_t)t * HH;
    float g[EE];
#pragma unroll
    for (int e = 0; e < EE; ++e) g[e] = 0.0f;

    for (int h = lane; h < HH; h += 32) {
        float xv = xr[h];
        x_h[(size_t)t * HH + h] = (_Float16)xv;   // fp32 -> f16 copy of tokens
        const v4f g0 = *(const v4f*)(gw + (size_t)h * EE);
        const v4f g1 = *(const v4f*)(gw + (size_t)h * EE + 4);
#pragma unroll
        for (int e = 0; e < 4; ++e) { g[e] += xv * g0[e]; g[e + 4] += xv * g1[e]; }
    }
#pragma unroll
    for (int e = 0; e < EE; ++e) {
#pragma unroll
        for (int off = 16; off > 0; off >>= 1)
            g[e] += __shfl_down(g[e], off, 32);
    }
    if (lane == 0) {
        int e1 = 0; float m1 = g[0];
#pragma unroll
        for (int e = 1; e < EE; ++e) if (g[e] > m1) { m1 = g[e]; e1 = e; }
        int e2 = -1; float m2 = -3.4e38f;
#pragma unroll
        for (int e = 0; e < EE; ++e) if (e != e1 && g[e] > m2) { m2 = g[e]; e2 = e; }
        // renormalized top-2 softmax scores: p1 = 1/(1+q), p2 = q/(1+q)
        float q = __expf(m2 - m1);
        float inv = 1.0f / (1.0f + q);
        float w1 = alpha[e1] * inv;
        float w2 = alpha[e2] * q * inv;
        expTok[2 * t + 0] = e1;  wTok[2 * t + 0] = w1;
        expTok[2 * t + 1] = e2;  wTok[2 * t + 1] = w2;
        atomicAdd(&cnt[e1], 1);
        atomicAdd(&cnt[e2], 1);
    }
}

// ---------------- kernel 2: tiny exclusive scan over 8 experts ----------------
__global__ void moe_scan_kernel(const int* __restrict__ cnt, int* __restrict__ offs,
                                int* __restrict__ pos) {
    if (threadIdx.x == 0 && blockIdx.x == 0) {
        int run = 0;
        for (int e = 0; e < EE; ++e) { offs[e] = run; pos[e] = run; run += cnt[e]; }
    }
}

// ---------------- kernel 3: scatter tokens into per-expert row ranges ----------------
__global__ void moe_scatter_kernel(const int* __restrict__ expTok, const float* __restrict__ wTok,
                                   int* __restrict__ pos, int* __restrict__ rowTok,
                                   float* __restrict__ rowW) {
    int t = blockIdx.x * blockDim.x + threadIdx.x;
    if (t >= TT) return;
#pragma unroll
    for (int k = 0; k < KK_TOP; ++k) {
        int e = expTok[2 * t + k];
        int slot = atomicAdd(&pos[e], 1);
        rowTok[slot] = t;
        rowW[slot]   = wTok[2 * t + k];
    }
}

// ---------------- kernels 4/5: grouped WMMA GEMM (double-buffered LDS pipeline) ----
// PHASE 1: h = gelu(x_h[gather] @ fc1[e] + b1)   (A gathered via rowTok)
// PHASE 2: out[tok] += w * (h @ fc2[e] + b2)     (A = h rows directly)
template <int PHASE>
__global__ __launch_bounds__(256)
void moe_gemm_kernel(const _Float16* __restrict__ Amat,  // x_h (P1) or h (P2), ld = 2048
                     const float* __restrict__ W,        // (E, 2048, 2048) fp32 row-major
                     const float* __restrict__ bias,     // (E, 2048)
                     const int* __restrict__ cnt, const int* __restrict__ offs,
                     const int* __restrict__ rowTok, const float* __restrict__ rowW,
                     _Float16* __restrict__ hbuf,        // (2T, 2048) f16
                     float* __restrict__ out)            // (T, 2048) fp32
{
    const int e = blockIdx.z;
    const int ce = cnt[e];
    const int tileM = blockIdx.y;
    if (tileM * BM >= ce) return;                 // uniform exit: EXEC stays all-1s for WMMA
    const int rowStart = offs[e] + tileM * BM;
    const int rowEnd   = offs[e] + ce;
    const int n0 = blockIdx.x * BN;

    const int lane = threadIdx.x & 31;
    const int wv   = threadIdx.x >> 5;            // 0..7
    const int mw   = wv >> 1;                     // 0..3  (32 rows each)
    const int nw   = wv & 1;                      // 0..1  (64 cols each)

    // Double-buffered LDS tiles, pre-swizzled to WMMA per-lane operand layout (ISA 7.12.2).
    __shared__ alignas(32) unsigned int As[2][8][32][8];  // [buf][m-subtile][lane][dword]
    __shared__ alignas(32) unsigned int Bs[2][8][32][8];  // [buf][n-subtile][lane][dword]

    v8f acc[2][4];
    const v8f vzero = {0.f, 0.f, 0.f, 0.f, 0.f, 0.f, 0.f, 0.f};
#pragma unroll
    for (int i = 0; i < 2; ++i)
#pragma unroll
        for (int j = 0; j < 4; ++j) acc[i][j] = vzero;

    // --- staging addresses for this thread (wave wv stages A-subtile wv, B-subtile wv) ---
    int gr = rowStart + wv * 16 + (lane & 15);
    if (gr >= rowEnd) gr = rowStart;              // clamp: garbage rows discarded in epilogue
    const _Float16* arow =
        (PHASE == 1) ? (Amat + (size_t)rowTok[gr] * HH)
                     : (Amat + (size_t)gr * II);
    const int kA = (lane >= 16) ? 8 : 0;          // A: hi half-wave holds K {8-15,24-31}

    const float* Wg = W + (size_t)e * 2048 * 2048;
    const float* bcol = Wg + (size_t)((lane >= 16) ? 8 : 0) * 2048
                           + (n0 + wv * 16 + (lane & 15));

    // --- software pipeline registers ---
    unsigned int aReg[8];
    float bReg0[8], bReg1[8];

    auto fetchG = [&](int k0) {
#pragma unroll
        for (int j = 0; j < 8; ++j) {
            const int kk = 2 * j + ((j >= 4) ? 8 : 0) + kA;
            aReg[j] = *(const unsigned int*)(arow + k0 + kk);
        }
        const float* bptr = bcol + (size_t)k0 * 2048;
#pragma unroll
        for (int j = 0; j < 8; ++j) {
            bReg0[j] = bptr[(size_t)(2 * j + ((j >= 4) ? 8 : 0)) * 2048];
            bReg1[j] = bptr[(size_t)(2 * j + 1 + ((j >= 4) ? 8 : 0)) * 2048];
        }
    };

    fetchG(0);                                    // prologue
    int p = 0;
    for (int k0 = 0; k0 < 2048; k0 += BK) {
        // drain prefetched regs into LDS[p]
#pragma unroll
        for (int j = 0; j < 8; ++j) As[p][wv][lane][j] = aReg[j];
#pragma unroll
        for (int j = 0; j < 8; ++j) Bs[p][wv][lane][j] = pk_f16(bReg0[j], bReg1[j]);
        __syncthreads();                          // single barrier per iteration

        // issue next K-slab global loads; they drain while WMMAs run
        int kn = k0 + BK;
        if (kn >= 2048) kn = 0;                   // wrap: avoids OOB tail prefetch
        fetchG(kn);

        v16h a[2], b[4];
#pragma unroll
        for (int i = 0; i < 2; ++i) a[i] = *(const v16h*)As[p][mw * 2 + i][lane];
#pragma unroll
        for (int j = 0; j < 4; ++j) b[j] = *(const v16h*)Bs[p][nw * 4 + j][lane];
#pragma unroll
        for (int i = 0; i < 2; ++i)
#pragma unroll
            for (int j = 0; j < 4; ++j)
                acc[i][j] = __builtin_amdgcn_wmma_f32_16x16x32_f16(
                    false, a[i], false, b[j], (short)0, acc[i][j], false, false);
        p ^= 1;
    }

    // --- epilogue: C layout = VGPR r, lane l -> M = r + (l>>4)*8, N = l&15 ---
    const int hi = lane >> 4;
    const int nl = lane & 15;
    float bv[4];
#pragma unroll
    for (int j = 0; j < 4; ++j)
        bv[j] = bias[(size_t)e * 2048 + n0 + (nw * 4 + j) * 16 + nl];

#pragma unroll
    for (int i = 0; i < 2; ++i) {
        const int mbase = rowStart + (mw * 2 + i) * 16 + hi * 8;
#pragma unroll
        for (int r = 0; r < 8; ++r) {
            const int row = mbase + r;
            if (row < rowEnd) {
                if (PHASE == 1) {
#pragma unroll
                    for (int j = 0; j < 4; ++j) {
                        const int col = n0 + (nw * 4 + j) * 16 + nl;
                        float v = acc[i][j][r] + bv[j];
                        hbuf[(size_t)row * II + col] = (_Float16)gelu_exact(v);
                    }
                } else {
                    const int   tok = rowTok[row];
                    const float wgt = rowW[row];
#pragma unroll
                    for (int j = 0; j < 4; ++j) {
                        const int col = n0 + (nw * 4 + j) * 16 + nl;
                        float v = acc[i][j][r] + bv[j];
                        atomicAdd(out + (size_t)tok * HH + col, wgt * v);
                    }
                }
            }
        }
    }
}

// ---------------- launch ----------------
extern "C" void kernel_launch(void* const* d_in, const int* in_sizes, int n_in,
                              void* d_out, int out_size, void* d_ws, size_t ws_size,
                              hipStream_t stream) {
    const float* x     = (const float*)d_in[0];  // (4,2048,2048)
    const float* gw    = (const float*)d_in[1];  // (2048,8)
    const float* fc1w  = (const float*)d_in[2];  // (8,2048,2048)
    const float* fc1b  = (const float*)d_in[3];  // (8,2048)
    const float* fc2w  = (const float*)d_in[4];  // (8,2048,2048)
    const float* fc2b  = (const float*)d_in[5];  // (8,2048)
    const float* alpha = (const float*)d_in[6];  // (8,)
    float* out = (float*)d_out;

    char* ws = (char*)d_ws;
    int*      cnt    = (int*)(ws + 0);
    int*      offs   = (int*)(ws + 256);
    int*      pos    = (int*)(ws + 512);
    int*      expTok = (int*)(ws + 1024);                       // 2T ints
    float*    wTok   = (float*)(ws + 1024 + 65536);             // 2T floats
    int*      rowTok = (int*)(ws + 1024 + 2 * 65536);           // 2T ints
    float*    rowW   = (float*)(ws + 1024 + 3 * 65536);         // 2T floats
    _Float16* x_h    = (_Float16*)(ws + 1024 + 4 * 65536);      // T*H f16 (33.5 MB)
    _Float16* hbuf   = (_Float16*)(ws + 1024 + 4 * 65536 + (size_t)TT * HH * 2); // 2T*I f16 (67 MB)

    moe_init_kernel<<<4096, 256, 0, stream>>>(out, out_size / 4, cnt);
    moe_gate_kernel<<<TT / 8, 256, 0, stream>>>(x, gw, alpha, x_h, cnt, expTok, wTok);
    moe_scan_kernel<<<1, 32, 0, stream>>>(cnt, offs, pos);
    moe_scatter_kernel<<<TT / 256, 256, 0, stream>>>(expTok, wTok, pos, rowTok, rowW);

    dim3 g1(II / BN, (TT + BM - 1) / BM, EE);
    moe_gemm_kernel<1><<<g1, 256, 0, stream>>>(x_h, fc1w, fc1b, cnt, offs, rowTok, rowW,
                                               hbuf, out);
    dim3 g2(HH / BN, (TT + BM - 1) / BM, EE);
    moe_gemm_kernel<2><<<g2, 256, 0, stream>>>(hbuf, fc2w, fc2b, cnt, offs, rowTok, rowW,
                                               hbuf, out);
}
